// FastSpeech2_68152541052958
// MI455X (gfx1250) — matrix operand order repaired and checked by reading, not compiled, not run on hardware
//
#include <hip/hip_runtime.h>
#include <cstdint>
#include <cstddef>

// ---------------------------------------------------------------------------
// FastSpeech2 forward on gfx1250 (MI455X): all GEMMs/convs via
// v_wmma_f32_16x16x32_bf16, attention fully LDS-resident (320KB/WGP).
// ---------------------------------------------------------------------------

namespace {
constexpr int B_ = 16, LS_ = 256, T_ = 1024, D_ = 256, H_ = 2, DFF_ = 1024, NMELS_ = 80;
}

typedef __attribute__((ext_vector_type(16))) __bf16 bf16x16;
typedef __attribute__((ext_vector_type(8)))  float  f32x8;

union FragBF16 {
  bf16x16 v;
  uint4 q[2];
  unsigned short u[16];
};

__device__ __forceinline__ unsigned short f2bf(float f) {
  unsigned int u = __float_as_uint(f);
  unsigned int r = 0x7FFFu + ((u >> 16) & 1u);   // round-to-nearest-even
  return (unsigned short)((u + r) >> 16);
}

__device__ __forceinline__ f32x8 wmma_bf16(const FragBF16& a, const FragBF16& b, f32x8 c) {
  // (neg_a, A, neg_b, B, c_mod, C, reuse_a, reuse_b)
  return __builtin_amdgcn_wmma_f32_16x16x32_bf16(false, a.v, false, b.v, (short)0, c, false, false);
}

// ---------------------------------------------------------------------------
// Generic GEMM: C[M,N] = A[M,K] @ W[K,N] + bias  (optional ReLU)
// CONV9 mode: A is x[B,Tlen,256], K = 9*256, tap = k/256 shifts time by tap-4
// Tlen is a power of two -> tShift/tMask instead of div/mod.
// Tile: 128x64x32, 256 threads (8 waves), wave = 32x32 (2x2 accums).
// ---------------------------------------------------------------------------
template <int CONV9>
__global__ __launch_bounds__(256) void gemm_bf16_kernel(
    const float* __restrict__ A, const float* __restrict__ W,
    const float* __restrict__ bias, float* __restrict__ C,
    int M, int N, int K, int tShift, int doRelu)
{
  __shared__ __align__(16) unsigned short As[128][32];   // [m][k]
  __shared__ __align__(16) unsigned short Bs[64][32];    // [n][k]
  const int tid  = threadIdx.x;
  const int lane = tid & 31;
  const int wave = tid >> 5;
  const int wm   = wave >> 1;      // 0..3
  const int wn   = wave & 1;       // 0..1
  const int mBase = blockIdx.y * 128;
  const int nBase = blockIdx.x * 64;
  const int hi = lane >> 4;        // 0/1
  const int ml = lane & 15;
  const int tMask = (1 << tShift) - 1;
  const int Tlen  = 1 << tShift;

  f32x8 acc[2][2];
  for (int i = 0; i < 2; ++i)
    for (int j = 0; j < 2; ++j)
      for (int e = 0; e < 8; ++e) acc[i][j][e] = 0.0f;

  // Per-thread staging coordinates are k-invariant: precompute rows.
  int aRow[16], aBT[16];   // aBT = b*Tlen (conv) base, aRow = global row, tt in low bits
  for (int i = 0; i < 16; ++i) {
    int idx = i * 256 + tid;           // 0..4095
    int r = idx >> 5;
    aRow[i] = mBase + r;
    aBT[i]  = (mBase + r) & ~tMask;    // b*Tlen
  }
  const int aCol = tid & 31;

  for (int k0 = 0; k0 < K; k0 += 32) {
    // --- stage A tile (f32 -> bf16) ---
    for (int i = 0; i < 16; ++i) {
      int row = aRow[i];
      float v = 0.0f;
      if (!CONV9) {
        if (row < M) v = A[(size_t)row * K + (k0 + aCol)];
      } else {
        int kk = k0 + aCol;
        int tap = kk >> 8;               // / 256
        int kc  = kk & 255;
        int tt = row & tMask;
        int ts = tt + tap - 4;           // pad = 4
        if (row < M && ((unsigned)ts < (unsigned)Tlen))
          v = A[((size_t)(aBT[i] + ts)) * 256 + kc];
      }
      As[(i * 256 + tid) >> 5][aCol] = f2bf(v);
    }
    // --- stage B tile transposed: Bs[n][k] ---
    for (int i = 0; i < 8; ++i) {
      int idx = i * 256 + tid;           // 0..2047
      int kk = idx >> 6, n = idx & 63;
      int col = nBase + n;
      float v = (col < N) ? W[(size_t)(k0 + kk) * N + col] : 0.0f;
      Bs[n][kk] = f2bf(v);
    }
    __syncthreads();

    FragBF16 fa[2], fb[2];
    for (int i = 0; i < 2; ++i) {
      int m = wm * 32 + i * 16 + ml;
      fa[i].q[0] = *(const uint4*)&As[m][hi * 8];
      fa[i].q[1] = *(const uint4*)&As[m][hi * 8 + 16];
    }
    for (int j = 0; j < 2; ++j) {
      int n = wn * 32 + j * 16 + ml;
      fb[j].q[0] = *(const uint4*)&Bs[n][hi * 8];
      fb[j].q[1] = *(const uint4*)&Bs[n][hi * 8 + 16];
    }
    for (int i = 0; i < 2; ++i)
      for (int j = 0; j < 2; ++j)
        acc[i][j] = wmma_bf16(fa[i], fb[j], acc[i][j]);
    __syncthreads();
  }

  for (int i = 0; i < 2; ++i)
    for (int j = 0; j < 2; ++j)
      for (int e = 0; e < 8; ++e) {
        int row = mBase + wm * 32 + i * 16 + hi * 8 + e;
        int col = nBase + wn * 32 + j * 16 + ml;
        if (row < M && col < N) {
          float v = acc[i][j][e] + bias[col];
          if (doRelu) v = fmaxf(v, 0.0f);
          C[(size_t)row * N + col] = v;
        }
      }
}

// ---------------------------------------------------------------------------
// Attention: per (b, h, 32-query block). Full score rows live in LDS.
// dk = 128. 128 threads (4 waves). Dynamic LDS: S[32][T] f32 + Q + K/V bf16.
// ---------------------------------------------------------------------------
__global__ __launch_bounds__(128) void attn_kernel(
    const float* __restrict__ Q, const float* __restrict__ Kb,
    const float* __restrict__ Vb, float* __restrict__ O,
    const int* __restrict__ mask, int Tlen)
{
  extern __shared__ char smem[];
  float* S = (float*)smem;                                        // [32][Tlen]
  unsigned short* Qs  = (unsigned short*)(smem + (size_t)32 * Tlen * 4); // [32][128]
  unsigned short* KVs = Qs + 32 * 128;                            // 64*128 staging
  const int tid = threadIdx.x;
  const int lane = tid & 31;
  const int wave = tid >> 5;
  const int hi = lane >> 4, ml = lane & 15;
  const int qb = blockIdx.x, h = blockIdx.y, b = blockIdx.z;
  const int qBase = qb * 32;
  const float* Qp = Q  + (size_t)b * Tlen * 256 + h * 128;
  const float* Kp = Kb + (size_t)b * Tlen * 256 + h * 128;
  const float* Vp = Vb + (size_t)b * Tlen * 256 + h * 128;

  for (int i = 0; i < 32; ++i) {
    int idx = i * 128 + tid;
    int r = idx >> 7, d = idx & 127;
    Qs[r * 128 + d] = f2bf(Qp[(size_t)(qBase + r) * 256 + d]);
  }
  __syncthreads();

  const float scale = 0.08838834764831845f;  // 1/sqrt(128)
  const int wq_ = wave & 1, wk_ = wave >> 1; // 2x2 wave grid over (q16, k32)

  // ---- Phase 1: S = Q K^T * scale, masked ----
  for (int kb0 = 0; kb0 < Tlen; kb0 += 64) {
    for (int i = 0; i < 64; ++i) {
      int idx = i * 128 + tid;
      int r = idx >> 7, d = idx & 127;
      KVs[r * 128 + d] = f2bf(Kp[(size_t)(kb0 + r) * 256 + d]);  // [kcol][d]
    }
    __syncthreads();
    f32x8 acc[2];
    for (int j = 0; j < 2; ++j)
      for (int e = 0; e < 8; ++e) acc[j][e] = 0.0f;
    for (int d0 = 0; d0 < 128; d0 += 32) {
      FragBF16 fa, fb[2];
      int m = wq_ * 16 + ml;
      fa.q[0] = *(const uint4*)&Qs[m * 128 + d0 + hi * 8];
      fa.q[1] = *(const uint4*)&Qs[m * 128 + d0 + hi * 8 + 16];
      for (int j = 0; j < 2; ++j) {
        int n = wk_ * 32 + j * 16 + ml;
        fb[j].q[0] = *(const uint4*)&KVs[n * 128 + d0 + hi * 8];
        fb[j].q[1] = *(const uint4*)&KVs[n * 128 + d0 + hi * 8 + 16];
        acc[j] = wmma_bf16(fa, fb[j], acc[j]);
      }
    }
    for (int j = 0; j < 2; ++j)
      for (int e = 0; e < 8; ++e) {
        int row = wq_ * 16 + hi * 8 + e;
        int col = kb0 + wk_ * 32 + j * 16 + ml;
        float sv = acc[j][e] * scale;
        if (mask[b * Tlen + col] == 0) sv = -__builtin_inff();
        S[(size_t)row * Tlen + col] = sv;
      }
    __syncthreads();
  }

  // ---- Phase 2: exact softmax per row (32 rows, one thread each) ----
  if (tid < 32) {
    float* row = S + (size_t)tid * Tlen;
    float m = -__builtin_inff();
    for (int t = 0; t < Tlen; ++t) m = fmaxf(m, row[t]);
    float s = 0.0f;
    for (int t = 0; t < Tlen; ++t) { float e = expf(row[t] - m); row[t] = e; s += e; }
    float inv = 1.0f / s;
    for (int t = 0; t < Tlen; ++t) row[t] *= inv;
  }
  __syncthreads();

  // ---- Phase 3: O = P V ----
  const int wm = wave & 1, wd = wave >> 1;   // (q16, d64)
  f32x8 oacc[4];
  for (int j = 0; j < 4; ++j)
    for (int e = 0; e < 8; ++e) oacc[j][e] = 0.0f;
  for (int kb0 = 0; kb0 < Tlen; kb0 += 64) {
    for (int i = 0; i < 64; ++i) {
      int idx = i * 128 + tid;
      int r = idx >> 7, d = idx & 127;
      KVs[d * 64 + r] = f2bf(Vp[(size_t)(kb0 + r) * 256 + d]);   // Vt[d][k]
    }
    __syncthreads();
    for (int s0 = 0; s0 < 2; ++s0) {
      FragBF16 fa;
      int m = wm * 16 + ml;
      const float* srow = S + (size_t)m * Tlen + kb0 + s0 * 32 + hi * 8;
      // 16 probs: [0..7] and [16..23] relative to srow; 32B-aligned float4 reads
      float4 p0 = *(const float4*)(srow + 0);
      float4 p1 = *(const float4*)(srow + 4);
      float4 p2 = *(const float4*)(srow + 16);
      float4 p3 = *(const float4*)(srow + 20);
      fa.u[0] = f2bf(p0.x);  fa.u[1] = f2bf(p0.y);
      fa.u[2] = f2bf(p0.z);  fa.u[3] = f2bf(p0.w);
      fa.u[4] = f2bf(p1.x);  fa.u[5] = f2bf(p1.y);
      fa.u[6] = f2bf(p1.z);  fa.u[7] = f2bf(p1.w);
      fa.u[8]  = f2bf(p2.x); fa.u[9]  = f2bf(p2.y);
      fa.u[10] = f2bf(p2.z); fa.u[11] = f2bf(p2.w);
      fa.u[12] = f2bf(p3.x); fa.u[13] = f2bf(p3.y);
      fa.u[14] = f2bf(p3.z); fa.u[15] = f2bf(p3.w);
      for (int j = 0; j < 4; ++j) {
        int n = wd * 64 + j * 16 + ml;
        FragBF16 fb;
        fb.q[0] = *(const uint4*)&KVs[n * 64 + s0 * 32 + hi * 8];
        fb.q[1] = *(const uint4*)&KVs[n * 64 + s0 * 32 + hi * 8 + 16];
        oacc[j] = wmma_bf16(fa, fb, oacc[j]);
      }
    }
    __syncthreads();
  }
  for (int j = 0; j < 4; ++j)
    for (int e = 0; e < 8; ++e) {
      int row = qBase + wm * 16 + hi * 8 + e;
      int col = wd * 64 + j * 16 + ml;
      O[((size_t)b * Tlen + row) * 256 + h * 128 + col] = oacc[j][e];
    }
}

// ---------------------------------------------------------------------------
// LayerNorm over D=256 (block-per-row). Optional residual add + npm mask.
// ---------------------------------------------------------------------------
__global__ __launch_bounds__(256) void ln_kernel(
    const float* __restrict__ y, const float* __restrict__ xres,
    const float* __restrict__ g, const float* __restrict__ bta,
    const int* __restrict__ mask, float* __restrict__ out)
{
  __shared__ float red[256];
  const int row = blockIdx.x;
  const int t = threadIdx.x;
  float v = y[(size_t)row * 256 + t];
  if (xres) v += xres[(size_t)row * 256 + t];
  red[t] = v;
  __syncthreads();
  for (int s = 128; s > 0; s >>= 1) { if (t < s) red[t] += red[t + s]; __syncthreads(); }
  float mean = red[0] * (1.0f / 256.0f);
  __syncthreads();
  float d = v - mean;
  red[t] = d * d;
  __syncthreads();
  for (int s = 128; s > 0; s >>= 1) { if (t < s) red[t] += red[t + s]; __syncthreads(); }
  float var = red[0] * (1.0f / 256.0f);
  float rs = rsqrtf(var + 1e-5f);
  float o = d * rs * g[t] + bta[t];
  if (mask && mask[row] == 0) o = 0.0f;
  out[(size_t)row * 256 + t] = o;
}

__global__ void addmask_kernel(const float* __restrict__ hsrc, const float* __restrict__ x,
                               const int* __restrict__ mask, float* __restrict__ out, int total)
{
  int idx = blockIdx.x * blockDim.x + threadIdx.x;
  if (idx >= total) return;
  int row = idx >> 8;
  float v = hsrc[idx] + x[idx];
  out[idx] = mask[row] ? v : 0.0f;
}

__global__ __launch_bounds__(256) void embed_enc_kernel(
    const int* __restrict__ seq, const int* __restrict__ pos,
    const float* __restrict__ wemb, const float* __restrict__ pemb,
    float* __restrict__ x, int* __restrict__ mask)
{
  int row = blockIdx.x;
  int d = threadIdx.x;
  int tok = seq[row];
  int pp = pos[row];
  x[(size_t)row * 256 + d] = wemb[(size_t)tok * 256 + d] + pemb[(size_t)pp * 256 + d];
  if (d == 0) mask[row] = (tok != 0) ? 1 : 0;
}

__global__ void cumsum_kernel(const int* __restrict__ dur, int* __restrict__ cum) {
  if (threadIdx.x != 0) return;
  int b = blockIdx.x;
  int s = 0;
  for (int i = 0; i < 256; ++i) { s += dur[b * 256 + i]; cum[b * 256 + i] = s; }
}

// Length regulator + pitch/energy bins + decoder positional embedding.
__global__ __launch_bounds__(256) void regulate_kernel(
    const float* __restrict__ xe, const int* __restrict__ cum,
    const float* __restrict__ pitch, const float* __restrict__ energy,
    const int* __restrict__ mel_pos, const float* __restrict__ pemb,
    const float* __restrict__ eemb, const float* __restrict__ posd,
    float* __restrict__ xd, int* __restrict__ mask)
{
  int row = blockIdx.x;           // b*1024 + t
  int b = row >> 10;
  int t = row & 1023;
  const int* c = cum + b * 256;
  int lo = 0, hiB = 256;          // searchsorted(cum, t, side='right')
  while (lo < hiB) { int mid = (lo + hiB) >> 1; if (c[mid] <= t) lo = mid + 1; else hiB = mid; }
  int idx = lo < 255 ? lo : 255;
  float pv = pitch[row], ev = energy[row];
  int pi = (int)ceilf(pv * 254.0f); pi = pi < 0 ? 0 : (pi > 255 ? 255 : pi);
  int ei = (int)ceilf(ev * 254.0f); ei = ei < 0 ? 0 : (ei > 255 ? 255 : ei);
  int mp = mel_pos[row];
  int d = threadIdx.x;
  xd[(size_t)row * 256 + d] = xe[((size_t)b * 256 + idx) * 256 + d]
      + pemb[(size_t)pi * 256 + d] + eemb[(size_t)ei * 256 + d]
      + posd[(size_t)mp * 256 + d];
  if (d == 0) mask[row] = (mp != 0) ? 1 : 0;
}

// ---------------------------------------------------------------------------
// Host orchestration
// ---------------------------------------------------------------------------
extern "C" void kernel_launch(void* const* d_in, const int* in_sizes, int n_in,
                              void* d_out, int out_size, void* d_ws, size_t ws_size,
                              hipStream_t stream)
{
  (void)in_sizes; (void)n_in; (void)out_size; (void)ws_size;
  // setup_inputs() insertion order, params flattened recursively in dict order
  const int*   src_seq = (const int*)d_in[0];
  const int*   src_pos = (const int*)d_in[1];
  const int*   mel_pos = (const int*)d_in[2];
  const int*   dur     = (const int*)d_in[3];
  const float* pitch   = (const float*)d_in[4];
  const float* energy  = (const float*)d_in[5];
  auto P = [&](int i) { return (const float*)d_in[i]; };
  const float* word_emb   = P(6);
  const float* pe_enc     = P(7);
  const float* pe_dec     = P(8);
  const float* pitch_emb  = P(9);
  const float* energy_emb = P(10);
  const float* mel_w      = P(11);
  const float* mel_b      = P(12);
  // layer leaves: wq,bq,wk,bk,wv,bv,wo,bo,ln1_g,ln1_b,w1,b1,w2,b2,ln2_g,ln2_b
  const int ENC_BASE = 13;
  const int DEC_BASE = 13 + 4 * 16;

  // workspace layout (bytes)
  char* base = (char*)d_ws;
  size_t off = 0;
  auto alloc = [&](size_t bytes) { void* p = base + off; off += (bytes + 255) & ~(size_t)255; return p; };
  float* xe  = (float*)alloc((size_t)B_ * LS_ * D_ * 4);    // encoder stream
  float* xd  = (float*)alloc((size_t)B_ * T_ * D_ * 4);     // decoder stream
  float* qb  = (float*)alloc((size_t)B_ * T_ * D_ * 4);
  float* kb  = (float*)alloc((size_t)B_ * T_ * D_ * 4);
  float* vb  = (float*)alloc((size_t)B_ * T_ * D_ * 4);
  float* yb  = (float*)alloc((size_t)B_ * T_ * D_ * 4);
  float* hff = (float*)alloc((size_t)B_ * T_ * DFF_ * 4);
  int* cum    = (int*)alloc((size_t)B_ * LS_ * 4);
  int* mask_e = (int*)alloc((size_t)B_ * LS_ * 4);
  int* mask_d = (int*)alloc((size_t)B_ * T_ * 4);

  auto gemm = [&](const float* A, const float* W, const float* bias, float* C,
                  int M, int N, int K, int tShift, int relu) {
    dim3 grid((N + 63) / 64, (M + 127) / 128);
    gemm_bf16_kernel<0><<<grid, 256, 0, stream>>>(A, W, bias, C, M, N, K, tShift, relu);
  };
  auto gemmConv9 = [&](const float* A, const float* W, const float* bias, float* C,
                       int M, int N, int K, int tShift, int relu) {
    dim3 grid((N + 63) / 64, (M + 127) / 128);
    gemm_bf16_kernel<1><<<grid, 256, 0, stream>>>(A, W, bias, C, M, N, K, tShift, relu);
  };

  auto runBlock = [&](float* x, int Tlen, int tShift, const int* mask, int pb) {
    int M = B_ * Tlen;
    const float *wq = P(pb + 0),  *bq  = P(pb + 1),  *wk = P(pb + 2),  *bk_ = P(pb + 3),
                *wv = P(pb + 4),  *bv  = P(pb + 5),  *wo = P(pb + 6),  *bo  = P(pb + 7),
                *g1 = P(pb + 8),  *be1 = P(pb + 9),  *w1 = P(pb + 10), *bc1 = P(pb + 11),
                *w2 = P(pb + 12), *bc2 = P(pb + 13), *g2 = P(pb + 14), *be2 = P(pb + 15);
    gemm(x, wq, bq,  qb, M, 256, 256, tShift, 0);
    gemm(x, wk, bk_, kb, M, 256, 256, tShift, 0);
    gemm(x, wv, bv,  vb, M, 256, 256, tShift, 0);
    size_t smem = (size_t)32 * Tlen * 4 + (size_t)32 * 128 * 2 + (size_t)64 * 128 * 2;
    attn_kernel<<<dim3(Tlen / 32, H_, B_), 128, smem, stream>>>(qb, kb, vb, yb, mask, Tlen);
    gemm(yb, wo, bo, qb, M, 256, 256, tShift, 0);           // o @ wo + bo -> qb (reuse)
    ln_kernel<<<M, 256, 0, stream>>>(qb, x, g1, be1, mask, x);          // x = LN(y + x)*npm
    ln_kernel<<<M, 256, 0, stream>>>(x, nullptr, g2, be2, nullptr, kb); // h = LN(x) -> kb
    gemmConv9(kb, w1, bc1, hff, M, DFF_, 9 * 256, tShift, 1);  // conv k=9 + ReLU
    gemm(hff, w2, bc2, yb, M, 256, DFF_, tShift, 0);           // conv k=1
    addmask_kernel<<<(M * 256 + 255) / 256, 256, 0, stream>>>(yb, x, mask, x, M * 256);
  };

  // ---- Encoder ----
  embed_enc_kernel<<<B_ * LS_, 256, 0, stream>>>(src_seq, src_pos, word_emb, pe_enc, xe, mask_e);
  for (int l = 0; l < 4; ++l) runBlock(xe, LS_, 8, mask_e, ENC_BASE + l * 16);

  // ---- Length regulator + variance adaptor + decoder pos emb ----
  cumsum_kernel<<<B_, 32, 0, stream>>>(dur, cum);
  regulate_kernel<<<B_ * T_, 256, 0, stream>>>(xe, cum, pitch, energy, mel_pos,
                                               pitch_emb, energy_emb, pe_dec, xd, mask_d);

  // ---- Decoder ----
  for (int l = 0; l < 4; ++l) runBlock(xd, T_, 10, mask_d, DEC_BASE + l * 16);

  // ---- Mel head ----
  gemm(xd, mel_w, mel_b, (float*)d_out, B_ * T_, NMELS_, 256, 10, 0);
}
